// WindowAttentionLayer_43267500540099
// MI455X (gfx1250) — matrix operand order repaired and checked by reading, not compiled, not run on hardware
//
#include <hip/hip_runtime.h>
#include <hip/hip_bf16.h>

// ---------------------------------------------------------------------------
// Sliding-window attention for MI455X (gfx1250, wave32).
// All GEMMs via v_wmma_f32_16x16x32_bf16 (bf16 operands, f32 accum).
//  K0: one-time f32->bf16 conversion of x and the weights (amortized; keeps
//      conversions OFF the GEMM critical path).
//  K1: fused QKV projection, 32x64 output tile per wave (2 A-frags x 4
//      B-frags = 8 WMMAs per K-step, pure b128-load + WMMA inner loop).
//      Q pre-scaled by 1/sqrt(64). V stored transposed [B,H,D,S].
//  K2: flash attention, 1 wave per (b,h,16-query tile); online softmax over
//      32-key blocks; P bounced through LDS (C-layout -> A-layout);
//      global_prefetch of the next K/V block.
// Fragment layouts (CDNA5 ISA):
//  A 16x32 bf16 : lane holds row M=lane%16, K = e + 8*((e>>3)+(lane>>4))
//  B 32x16 bf16 : lane holds col N=lane%16, same K as A; code uses K = e + 16*(lane>>4) for both
//  C 16x16 f32  : elem r -> M = r + 8*(lane>>4), N = lane%16
// ---------------------------------------------------------------------------

typedef __bf16 v16bf __attribute__((ext_vector_type(16)));
typedef __bf16 v8bf  __attribute__((ext_vector_type(8)));
typedef float  v8f   __attribute__((ext_vector_type(8)));

#define BB   2
#define SS   2048
#define DIMM 1024
#define NH   16
#define HD   64
#define WIN  256

__device__ __forceinline__ v16bf cat16(v8bf lo, v8bf hi) {
  return __builtin_shufflevector(lo, hi, 0, 1, 2, 3, 4, 5, 6, 7,
                                 8, 9, 10, 11, 12, 13, 14, 15);
}
// A-fragment chunk pair: elements 0..7 at p[0..7], 8..15 at p[8..15].
__device__ __forceinline__ v16bf load_a_frag_bf16(const __bf16* p) {
  v8bf lo = *(const v8bf*)(p);
  v8bf hi = *(const v8bf*)(p + 8);
  return cat16(lo, hi);
}
// 16 contiguous bf16 -> one B fragment (element order == memory order).
__device__ __forceinline__ v16bf load_b_frag_bf16(const __bf16* p) {
  return *(const v16bf*)(p);
}

// ---------------------------------------------------------------------------
// Kernel 0: elementwise f32 -> bf16, 8 elements per thread (v_cvt_pk pairs).
// n must be a multiple of 8 (all tensors here are multiples of 1024).
// ---------------------------------------------------------------------------
__global__ __launch_bounds__(256)
void cvt_f32_bf16_kernel(const float* __restrict__ src,
                         __bf16* __restrict__ dst, int n) {
  const int i = (blockIdx.x * 256 + threadIdx.x) * 8;
  if (i < n) {
    v8f f = *(const v8f*)(src + i);
    *(v8bf*)(dst + i) = __builtin_convertvector(f, v8bf);
  }
}

// ---------------------------------------------------------------------------
// Kernel 1: QKV projection. grid = ((B*S/32)*(DIM/64), 3), block = 32.
// out[s, o] = sum_i x[s,i]*W[o,i] + b[o];  32x64 tile per wave.
// ---------------------------------------------------------------------------
__global__ __launch_bounds__(32)
void qkv_proj_kernel(const __bf16* __restrict__ xb,
                     const __bf16* __restrict__ Wqb,
                     const __bf16* __restrict__ Wkb,
                     const __bf16* __restrict__ Wvb,
                     const float* __restrict__ bq,
                     const float* __restrict__ bk,
                     const float* __restrict__ bv,
                     __bf16* __restrict__ qo,
                     __bf16* __restrict__ ko,
                     __bf16* __restrict__ vo) {
  const int lane  = threadIdx.x;
  const int lhalf = lane >> 4;
  const int lmod  = lane & 15;
  const int mat   = blockIdx.y;                    // 0=Q 1=K 2=V

  const int colTiles = DIMM / 64;                  // 16
  const int row0 = (blockIdx.x / colTiles) * 32;   // token row in [0,4096)
  const int col0 = (blockIdx.x % colTiles) * 64;   // output feature base

  const __bf16* W   = (mat == 0) ? Wqb : (mat == 1) ? Wkb : Wvb;
  const float* bias = (mat == 0) ? bq : (mat == 1) ? bk : bv;

  v8f acc[2][4];
  #pragma unroll
  for (int m = 0; m < 2; ++m)
    #pragma unroll
    for (int t = 0; t < 4; ++t)
      acc[m][t] = (v8f){0.f,0.f,0.f,0.f,0.f,0.f,0.f,0.f};

  const __bf16* xrow0 = xb + (size_t)(row0 + lmod) * DIMM;        // A rows
  const __bf16* xrow1 = xrow0 + 16 * (size_t)DIMM;
  const __bf16* wrow0 = W + (size_t)(col0 + lmod) * DIMM;         // B col bases
  const __bf16* wrow1 = wrow0 + 16 * (size_t)DIMM;
  const __bf16* wrow2 = wrow1 + 16 * (size_t)DIMM;
  const __bf16* wrow3 = wrow2 + 16 * (size_t)DIMM;
  const __bf16* wrows[4] = {wrow0, wrow1, wrow2, wrow3};

  for (int k0 = 0; k0 < DIMM; k0 += 32) {
    const v16bf a0 = load_a_frag_bf16(xrow0 + k0 + 16 * lhalf);
    const v16bf a1 = load_a_frag_bf16(xrow1 + k0 + 16 * lhalf);
    #pragma unroll
    for (int t = 0; t < 4; ++t) {
      const v16bf bfrag = load_b_frag_bf16(wrows[t] + k0 + 16 * lhalf);
      acc[0][t] = __builtin_amdgcn_wmma_f32_16x16x32_bf16(false, a0, false, bfrag,
                                                          (short)0, acc[0][t],
                                                          false, false);
      acc[1][t] = __builtin_amdgcn_wmma_f32_16x16x32_bf16(false, a1, false, bfrag,
                                                          (short)0, acc[1][t],
                                                          false, false);
    }
  }

  const float qscale = (mat == 0) ? 0.125f : 1.0f;   // fold 1/sqrt(64) into Q

  #pragma unroll
  for (int mrt = 0; mrt < 2; ++mrt) {
    #pragma unroll
    for (int t = 0; t < 4; ++t) {
      const int colg = col0 + t * 16 + lmod;
      const float bval = bias[colg];
      const int h = colg / HD, d = colg % HD;
      #pragma unroll
      for (int r = 0; r < 8; ++r) {
        const int rowg = row0 + 16 * mrt + r + 8 * lhalf;
        const int b = rowg / SS, s = rowg % SS;
        const __bf16 us = (__bf16)((acc[mrt][t][r] + bval) * qscale);
        if (mat == 2) {
          vo[(((size_t)b * NH + h) * HD + d) * SS + s] = us;   // V: [B,H,D,S]
        } else {
          __bf16* o = (mat == 0) ? qo : ko;
          o[(((size_t)b * NH + h) * SS + s) * HD + d] = us;    // Q/K: [B,H,S,D]
        }
      }
    }
  }
}

// ---------------------------------------------------------------------------
// Kernel 2: flash attention. grid = B*H*(S/16), block = 32 (1 wave).
// Mask: key j visible to query i iff (j - i) < WIN.
// ---------------------------------------------------------------------------
__global__ __launch_bounds__(32)
void attn_kernel(const __bf16* __restrict__ q,
                 const __bf16* __restrict__ k,
                 const __bf16* __restrict__ vt,
                 float* __restrict__ out) {
  __shared__ __bf16 Pbuf[16 * 32];   // P tile [16 q][32 keys], row-major

  const int lane  = threadIdx.x;
  const int lhalf = lane >> 4;
  const int lmod  = lane & 15;

  const int id = blockIdx.x;
  const int qt = id & 127;           // S/16 = 128
  const int h  = (id >> 7) & 15;
  const int b  = id >> 11;
  const int q0 = qt * 16;

  // Q A-fragments for d 0..31 and 32..63.
  const __bf16* qrow =
      q + (((size_t)b * NH + h) * SS + q0 + lmod) * HD + 16 * lhalf;
  const v16bf qa0 = load_a_frag_bf16(qrow);
  const v16bf qa1 = load_a_frag_bf16(qrow + 32);

  v8f o0 = (v8f){0.f,0.f,0.f,0.f,0.f,0.f,0.f,0.f};
  v8f o1 = o0, o2 = o0, o3 = o0;
  float mrow[8], lrow[8];
  #pragma unroll
  for (int r = 0; r < 8; ++r) { mrow[r] = -__builtin_inff(); lrow[r] = 0.f; }

  int kmax = q0 + 15 + WIN + 1;      // exclusive bound of useful keys
  if (kmax > SS) kmax = SS;
  const int nb = (kmax + 31) / 32;

  const __bf16* kbase = k  + (((size_t)b * NH + h) * SS) * HD;
  const __bf16* vbase = vt + (((size_t)b * NH + h) * HD) * SS;

  for (int kb = 0; kb < nb; ++kb) {
    const int key0 = kb * 32;

    // Prefetch next key block (K rows + Vt columns) into cache.
    if (kb + 1 < nb) {
      const int nkey = key0 + 32;
      __builtin_prefetch(kbase + (size_t)(nkey + lane) * HD, 0, 0);
      __builtin_prefetch(vbase + (size_t)(2 * lane) * SS + nkey, 0, 0);
      __builtin_prefetch(vbase + (size_t)(2 * lane + 1) * SS + nkey, 0, 0);
    }

    // ---- scores: two 16x16 C tiles (keys key0+0..15 / +16..31) ----
    v8f s0, s1;
    #pragma unroll
    for (int kg = 0; kg < 2; ++kg) {
      const __bf16* krow =
          kbase + (size_t)(key0 + kg * 16 + lmod) * HD + 16 * lhalf;
      const v16bf kb0 = load_b_frag_bf16(krow);        // d 0..31
      const v16bf kb1 = load_b_frag_bf16(krow + 32);   // d 32..63
      v8f sc = (v8f){0.f,0.f,0.f,0.f,0.f,0.f,0.f,0.f};
      sc = __builtin_amdgcn_wmma_f32_16x16x32_bf16(false, qa0, false, kb0,
                                                   (short)0, sc, false, false);
      sc = __builtin_amdgcn_wmma_f32_16x16x32_bf16(false, qa1, false, kb1,
                                                   (short)0, sc, false, false);
      if (kg == 0) s0 = sc; else s1 = sc;
    }

    // ---- mask boundary blocks ----
    if (key0 + 31 - q0 >= WIN) {
      #pragma unroll
      for (int r = 0; r < 8; ++r) {
        const int i = q0 + r + 8 * lhalf;
        if (key0 + lmod      - i >= WIN) s0[r] = -__builtin_inff();
        if (key0 + 16 + lmod - i >= WIN) s1[r] = -__builtin_inff();
      }
    }

    // ---- online softmax: row stats across 16-lane half-wave ----
    #pragma unroll
    for (int r = 0; r < 8; ++r) {
      float mx = fmaxf(s0[r], s1[r]);
      mx = fmaxf(mx, __shfl_xor(mx, 1, 32));
      mx = fmaxf(mx, __shfl_xor(mx, 2, 32));
      mx = fmaxf(mx, __shfl_xor(mx, 4, 32));
      mx = fmaxf(mx, __shfl_xor(mx, 8, 32));
      const float mnew  = fmaxf(mrow[r], mx);
      const float scale = __expf(mrow[r] - mnew);
      const float p0 = __expf(s0[r] - mnew);
      const float p1 = __expf(s1[r] - mnew);
      float sum = p0 + p1;
      sum += __shfl_xor(sum, 1, 32);
      sum += __shfl_xor(sum, 2, 32);
      sum += __shfl_xor(sum, 4, 32);
      sum += __shfl_xor(sum, 8, 32);
      lrow[r] = lrow[r] * scale + sum;
      mrow[r] = mnew;
      o0[r] *= scale; o1[r] *= scale; o2[r] *= scale; o3[r] *= scale;
      const int m = r + 8 * lhalf;          // C-layout row
      Pbuf[m * 32 + lmod]      = (__bf16)p0;
      Pbuf[m * 32 + 16 + lmod] = (__bf16)p1;
    }

    __syncthreads();
    // Reload P in A-layout: row M = lmod, K = e + 16*lhalf.
    const v16bf pa = load_a_frag_bf16(&Pbuf[lmod * 32 + 16 * lhalf]);
    __syncthreads();

    // ---- O += P * V : 4 WMMAs across the 64-wide head dim ----
    #pragma unroll
    for (int t = 0; t < 4; ++t) {
      const __bf16* vrow =
          vbase + (size_t)(t * 16 + lmod) * SS + key0 + 16 * lhalf;
      const v16bf vb = load_b_frag_bf16(vrow);
      v8f& ot = (t == 0) ? o0 : (t == 1) ? o1 : (t == 2) ? o2 : o3;
      ot = __builtin_amdgcn_wmma_f32_16x16x32_bf16(false, pa, false, vb,
                                                   (short)0, ot, false, false);
    }
  }

  // ---- normalize, write out [B,S,H,D] f32 ----
  #pragma unroll
  for (int r = 0; r < 8; ++r) {
    const int m = r + 8 * lhalf;
    const float inv = 1.0f / lrow[r];
    const size_t base = (((size_t)b * SS + q0 + m) * NH + h) * HD;
    out[base +  0 + lmod] = o0[r] * inv;
    out[base + 16 + lmod] = o1[r] * inv;
    out[base + 32 + lmod] = o2[r] * inv;
    out[base + 48 + lmod] = o3[r] * inv;
  }
}

// ---------------------------------------------------------------------------
extern "C" void kernel_launch(void* const* d_in, const int* in_sizes, int n_in,
                              void* d_out, int out_size, void* d_ws, size_t ws_size,
                              hipStream_t stream) {
  const float* x  = (const float*)d_in[0];
  const float* Wq = (const float*)d_in[1];
  const float* bq = (const float*)d_in[2];
  const float* Wk = (const float*)d_in[3];
  const float* bk = (const float*)d_in[4];
  const float* Wv = (const float*)d_in[5];
  const float* bv = (const float*)d_in[6];
  float* out = (float*)d_out;

  const size_t nx = (size_t)BB * SS * DIMM;    // 4M elements
  const size_t nw = (size_t)DIMM * DIMM;       // 1M elements

  __bf16* qws = (__bf16*)d_ws;                 //  8 MB
  __bf16* kws = qws + nx;                      //  8 MB
  __bf16* vws = kws + nx;                      //  8 MB
  __bf16* xbf = vws + nx;                      //  8 MB
  __bf16* wqb = xbf + nx;                      //  2 MB
  __bf16* wkb = wqb + nw;                      //  2 MB
  __bf16* wvb = wkb + nw;                      //  2 MB  (38 MB total)

  // K0: f32 -> bf16 conversions (one-time, off the GEMM critical path).
  cvt_f32_bf16_kernel<<<dim3((nx / 8 + 255) / 256), 256, 0, stream>>>(x, xbf, (int)nx);
  cvt_f32_bf16_kernel<<<dim3((nw / 8 + 255) / 256), 256, 0, stream>>>(Wq, wqb, (int)nw);
  cvt_f32_bf16_kernel<<<dim3((nw / 8 + 255) / 256), 256, 0, stream>>>(Wk, wkb, (int)nw);
  cvt_f32_bf16_kernel<<<dim3((nw / 8 + 255) / 256), 256, 0, stream>>>(Wv, wvb, (int)nw);

  // K1: QKV projection (32x64 tile per wave).
  dim3 gproj((BB * SS / 32) * (DIMM / 64), 3);
  qkv_proj_kernel<<<gproj, 32, 0, stream>>>(xbf, wqb, wkb, wvb, bq, bk, bv,
                                            qws, kws, vws);

  // K2: flash attention.
  dim3 gattn(BB * NH * (SS / 16));
  attn_kernel<<<gattn, 32, 0, stream>>>(qws, kws, vws, out);
}
